// LocalAttention_88003879895401
// MI455X (gfx1250) — matrix operand order repaired and checked by reading, not compiled, not run on hardware
//
#include <hip/hip_runtime.h>
#include <hip/hip_bf16.h>

typedef _Float16 half_t;
typedef __attribute__((ext_vector_type(16))) _Float16 v16h;
typedef __attribute__((ext_vector_type(8)))  _Float16 v8h;
typedef __attribute__((ext_vector_type(4)))  _Float16 v4h;
typedef __attribute__((ext_vector_type(8)))  float    v8f;

#define BATCH  2
#define SEQ    2048
#define DMODEL 1024
#define NHEADS 16
#define HDIM   64
#define WIN    128
#define NTOK   (BATCH * SEQ)          // 4096
#define NEG_BIG  (-1.0e30f)
#define NEG_THR  (-1.0e29f)

__device__ __forceinline__ v8f wmma32(v16h a, v16h b, v8f c) {
  // D = A(16x32 f16) * B(32x16 f16) + C(16x16 f32)
  return __builtin_amdgcn_wmma_f32_16x16x32_f16(
      /*neg_a=*/false, a, /*neg_b=*/false, b,
      /*c_mod=*/(short)0, c, /*reuse_a=*/false, /*reuse_b=*/false);
}

// A fragment: lane holds row (lane%16); elems 0..15 = contiguous
// row[2*c0..2*c0+16), c0 = 8*(lane/16).  (matches contiguous B order)
__device__ __forceinline__ v16h load_afrag(const half_t* row, int c0) {
  v8h lo = *(const v8h*)(row + 2 * c0);
  v8h hi = *(const v8h*)(row + 2 * c0 + 8);
  return __builtin_shufflevector(lo, hi,
      0,1,2,3,4,5,6,7,8,9,10,11,12,13,14,15);
}

// B fragment: lane holds column (lane%16); elems 0..15 = contiguous
// K-range [16*(lane/16), +16) of that column.  (contiguous k order, as in load_afrag)
__device__ __forceinline__ v16h load_bfrag(const half_t* col_at_kbase) {
  return *(const v16h*)col_at_kbase;
}

__device__ __forceinline__ int clampi(int v, int lo, int hi) {
  return v < lo ? lo : (v > hi ? hi : v);
}

// ---- CDNA5 async Global->LDS copy (ASYNCcnt path, CDNA5 ISA) -------
// Per lane: LDS[lds_off] = MEM[gaddr], 16 bytes.  VDST VGPR holds the LDS
// byte offset, VADDR the 64-bit global address; "off" = SADDR NULL.
__device__ __forceinline__ void async_cp_b128(unsigned lds_off, const void* gaddr) {
  asm volatile("global_load_async_to_lds_b128 %0, %1, off"
               :: "v"(lds_off), "v"(gaddr) : "memory");
}
__device__ __forceinline__ void wait_async_le1() {
  asm volatile("s_wait_asynccnt 0x1" ::: "memory");
}
__device__ __forceinline__ void wait_async_le0() {
  asm volatile("s_wait_asynccnt 0x0" ::: "memory");
}

// ---------------------------------------------------------------- converts
__global__ __launch_bounds__(256) void cvt_f32_to_f16x4(
    const float* __restrict__ s, half_t* __restrict__ d, int n4) {
  int i = blockIdx.x * 256 + threadIdx.x;
  if (i < n4) {
    float4 v = ((const float4*)s)[i];
    v4h o = { (half_t)v.x, (half_t)v.y, (half_t)v.z, (half_t)v.w };
    ((v4h*)d)[i] = o;
  }
}

// ------------------------------------------------------------- QKV GEMM
// Block = 8 waves covering a 128-row M slab x one 64-wide N tile.
// The 64x32 weight tile is staged once per block per k-step through a
// double-buffered LDS tile with async copies (8x less L2 traffic than
// per-wave fragment loads); A fragments go straight to VGPRs.  The k-loop
// is unrolled by 2 so each half addresses a fixed LDS buffer (keeps the
// WMMA accumulators pinned; no per-iteration buffer-select branch).
__global__ __launch_bounds__(256) void qkv_gemm_kernel(
    const half_t* __restrict__ xh, const half_t* __restrict__ wh,
    const float* __restrict__ bias,
    half_t* __restrict__ Qb, half_t* __restrict__ Kb, half_t* __restrict__ Vb) {
  __shared__ __align__(128) half_t Wl[2][64 * 32];   // 2 x 4KB
  const int tid  = threadIdx.x;
  const int lane = tid & 31;
  const int wave = tid >> 5;
  const int mtb = blockIdx.x / 48;                   // 32 M-slabs
  const int nt  = blockIdx.x % 48;                   // 48 N tiles (64-wide)
  const int mt  = mtb * 8 + wave;
  const int hs = lane >> 4, lr = lane & 15, c0 = hs * 8;

  const half_t* arow = xh + (size_t)(mt * 16 + lr) * DMODEL;
  // cooperative staging: thread t copies 16B chunk (t&3) of W row (t>>2)
  const half_t* wsrc = wh + (size_t)(nt * 64 + (tid >> 2)) * DMODEL + (tid & 3) * 8;
  const unsigned lds0   = (unsigned)(uintptr_t)&Wl[0][0];
  const unsigned ldsoff = lds0 + (unsigned)((tid >> 2) * 64 + (tid & 3) * 16);

  v8f acc[4];
  v8f z = {0.f,0.f,0.f,0.f,0.f,0.f,0.f,0.f};
  acc[0] = z; acc[1] = z; acc[2] = z; acc[3] = z;

  async_cp_b128(ldsoff, wsrc);                       // ks=0 -> buf0
#pragma unroll 1
  for (int ks = 0; ks < 32; ks += 2) {
    // ---- even step: consume buf0, prefetch ks+1 -> buf1 (always valid)
    async_cp_b128(ldsoff + 4096, wsrc + (ks + 1) * 32);
    wait_async_le1();                                // buf0 landed
    __syncthreads();                                 // all waves' copies visible
    __builtin_prefetch(arow + ks * 32 + 128, 0, 1);  // global_prefetch_b8
    {
      v16h af = load_afrag(arow + ks * 32, c0);
#pragma unroll
      for (int c = 0; c < 4; ++c) {
        v16h bf = *(const v16h*)&Wl[0][(c * 16 + lr) * 32 + 16 * hs];
        acc[c] = wmma32(af, bf, acc[c]);
      }
    }
    __syncthreads();                                 // done reading buf0

    // ---- odd step: consume buf1, prefetch ks+2 -> buf0 (except last)
    if (ks + 2 < 32) {
      async_cp_b128(ldsoff, wsrc + (ks + 2) * 32);
      wait_async_le1();                              // buf1 landed
    } else {
      wait_async_le0();
    }
    __syncthreads();
    {
      v16h af = load_afrag(arow + ks * 32 + 32, c0);
#pragma unroll
      for (int c = 0; c < 4; ++c) {
        v16h bf = *(const v16h*)&Wl[1][(c * 16 + lr) * 32 + 16 * hs];
        acc[c] = wmma32(af, bf, acc[c]);
      }
    }
    __syncthreads();                                 // done reading buf1
  }

#pragma unroll
  for (int c = 0; c < 4; ++c) {
    int col = nt * 64 + c * 16 + lr;
    float bval = bias[col];
#pragma unroll
    for (int r = 0; r < 8; ++r) {
      int token = mt * 16 + r + 8 * hs;
      int bb = token >> 11, ss = token & (SEQ - 1);
      half_t hv = (half_t)(acc[c][r] + bval);
      if (col < DMODEL) {
        int hh = col >> 6, dh = col & 63;
        Qb[(((size_t)(bb * NHEADS + hh) * SEQ + ss) << 6) + dh] = hv;
      } else if (col < 2 * DMODEL) {
        int cc = col - DMODEL; int hh = cc >> 6, dh = cc & 63;
        Kb[(((size_t)(bb * NHEADS + hh) * SEQ + ss) << 6) + dh] = hv;
      } else {
        int cc = col - 2 * DMODEL; int hh = cc >> 6, dh = cc & 63;
        Vb[((size_t)(bb * NHEADS + hh) * HDIM + dh) * SEQ + ss] = hv;   // V^T
      }
    }
  }
}

// ------------------------------------------------------- windowed attention
// One wave per (b, h, 16-query tile). Streams 9 key-blocks of 32 with
// online softmax; P re-shaped through a per-wave LDS tile for the PV WMMA.
__global__ __launch_bounds__(256) void attn_kernel(
    const half_t* __restrict__ Qb, const half_t* __restrict__ Kb,
    const half_t* __restrict__ Vb, half_t* __restrict__ Ob) {
  __shared__ __align__(128) half_t Pl[8][16 * 32];   // 1KB per wave
  const int lane = threadIdx.x & 31;
  const int wave = threadIdx.x >> 5;
  const int task = blockIdx.x * 8 + wave;            // 2*16*128 tasks
  const int qt = task & 127;
  const int h  = (task >> 7) & 15;
  const int b  = task >> 11;
  const int q0 = qt * 16;
  const int hs = lane >> 4, lr = lane & 15, c0 = hs * 8;

  const half_t* Qh = Qb + ((size_t)(b * NHEADS + h) * SEQ) * HDIM;
  const half_t* Kh = Kb + ((size_t)(b * NHEADS + h) * SEQ) * HDIM;
  const half_t* Vh = Vb + ((size_t)(b * NHEADS + h) * HDIM) * SEQ;   // [64][S]
  half_t* P = &Pl[wave][0];

  const half_t* qrow = Qh + (size_t)(q0 + lr) * HDIM;
  v16h qf0 = load_afrag(qrow, c0);                   // dh 0..31
  v16h qf1 = load_afrag(qrow + 32, c0);              // dh 32..63

  v8f z = {0.f,0.f,0.f,0.f,0.f,0.f,0.f,0.f};
  v8f o[4]; o[0] = z; o[1] = z; o[2] = z; o[3] = z;
  float mrun[8], lrun[8];
#pragma unroll
  for (int r = 0; r < 8; ++r) { mrun[r] = NEG_BIG; lrun[r] = 0.f; }

  const float scale = 0.125f;                        // 1/sqrt(64)

  for (int blk = 0; blk < 9; ++blk) {                // uniform trip count
    int j0 = q0 - WIN + blk * 32;
    v8f s[2];
#pragma unroll
    for (int sub = 0; sub < 2; ++sub) {              // 16 keys each
      int j  = j0 + sub * 16 + lr;
      int jc = clampi(j, 0, SEQ - 1);
      const half_t* kcol = Kh + (size_t)jc * HDIM + 16 * hs;
      v8f a = z;
      a = wmma32(qf0, load_bfrag(kcol), a);
      a = wmma32(qf1, load_bfrag(kcol + 32), a);
      s[sub] = a;
    }
    // mask + scale
#pragma unroll
    for (int sub = 0; sub < 2; ++sub) {
      int j = j0 + sub * 16 + lr;
      bool jin = (j >= 0) && (j < SEQ);
#pragma unroll
      for (int r = 0; r < 8; ++r) {
        int i = q0 + r + 8 * hs;
        bool ok = jin && (j >= i - WIN) && (j <= i + WIN);
        s[sub][r] = ok ? s[sub][r] * scale : NEG_BIG;
      }
    }
    // online softmax update (row = 16 lanes of one half-wave)
    float mnew[8], sc[8];
#pragma unroll
    for (int r = 0; r < 8; ++r) {
      float mx = fmaxf(s[0][r], s[1][r]);
#pragma unroll
      for (int off = 1; off < 16; off <<= 1)
        mx = fmaxf(mx, __shfl_xor(mx, off, 32));
      mnew[r] = fmaxf(mrun[r], mx);
      sc[r] = (mrun[r] > NEG_THR) ? __expf(mrun[r] - mnew[r]) : 0.f;
    }
#pragma unroll
    for (int sub = 0; sub < 2; ++sub)
#pragma unroll
      for (int r = 0; r < 8; ++r)
        s[sub][r] = (s[sub][r] > NEG_THR) ? __expf(s[sub][r] - mnew[r]) : 0.f;
#pragma unroll
    for (int r = 0; r < 8; ++r) {
      float rs = s[0][r] + s[1][r];
#pragma unroll
      for (int off = 1; off < 16; off <<= 1)
        rs += __shfl_xor(rs, off, 32);
      lrun[r] = lrun[r] * sc[r] + rs;
      mrun[r] = mnew[r];
    }
#pragma unroll
    for (int c = 0; c < 4; ++c)
#pragma unroll
      for (int r = 0; r < 8; ++r)
        o[c][r] *= sc[r];

    // P (C-layout) -> LDS row-major [16 x 32]
#pragma unroll
    for (int sub = 0; sub < 2; ++sub)
#pragma unroll
      for (int r = 0; r < 8; ++r)
        P[(r + 8 * hs) * 32 + sub * 16 + lr] = (half_t)s[sub][r];
    __syncthreads();

    // PV: A = P(16x32), B = V^T columns (dh fixed per lane, keys contiguous)
    v16h pf = load_afrag(P + lr * 32, c0);
    bool vin = (j0 >= 0) && (j0 + 31 < SEQ);
#pragma unroll
    for (int c = 0; c < 4; ++c) {
      int dh = c * 16 + lr;
      const half_t* vrow = Vh + (size_t)dh * SEQ;
      v16h vf;
      if (vin) {
        vf = load_bfrag(vrow + j0 + 16 * hs);
      } else {
#pragma unroll
        for (int e = 0; e < 16; ++e)
          vf[e] = vrow[clampi(j0 + 16 * hs + e, 0, SEQ - 1)];
      }
      o[c] = wmma32(pf, vf, o[c]);
    }
    __syncthreads();
  }

  // finalize: O / l, write f16 [token, d] for the output projection
#pragma unroll
  for (int c = 0; c < 4; ++c) {
#pragma unroll
    for (int r = 0; r < 8; ++r) {
      float denom = lrun[r];
      float val = (denom > 0.f) ? o[c][r] / denom : 0.f;
      int i = q0 + r + 8 * hs;
      int dh = c * 16 + lr;
      Ob[((size_t)(b * SEQ + i)) * DMODEL + h * HDIM + dh] = (half_t)val;
    }
  }
}

// --------------------------------------------------------- output projection
// Same block-cooperative async-LDS staging as the QKV GEMM.
__global__ __launch_bounds__(256) void oproj_kernel(
    const half_t* __restrict__ ah, const half_t* __restrict__ wh,
    const float* __restrict__ bias, float* __restrict__ out) {
  __shared__ __align__(128) half_t Wl[2][64 * 32];   // 2 x 4KB
  const int tid  = threadIdx.x;
  const int lane = tid & 31;
  const int wave = tid >> 5;
  const int mtb = blockIdx.x / 16;                   // 32 M-slabs
  const int nt  = blockIdx.x % 16;                   // 16 N tiles (64-wide)
  const int mt  = mtb * 8 + wave;
  const int hs = lane >> 4, lr = lane & 15, c0 = hs * 8;

  const half_t* arow = ah + (size_t)(mt * 16 + lr) * DMODEL;
  const half_t* wsrc = wh + (size_t)(nt * 64 + (tid >> 2)) * DMODEL + (tid & 3) * 8;
  const unsigned lds0   = (unsigned)(uintptr_t)&Wl[0][0];
  const unsigned ldsoff = lds0 + (unsigned)((tid >> 2) * 64 + (tid & 3) * 16);

  v8f acc[4];
  v8f z = {0.f,0.f,0.f,0.f,0.f,0.f,0.f,0.f};
  acc[0] = z; acc[1] = z; acc[2] = z; acc[3] = z;

  async_cp_b128(ldsoff, wsrc);                       // ks=0 -> buf0
#pragma unroll 1
  for (int ks = 0; ks < 32; ks += 2) {
    // even step: consume buf0, prefetch ks+1 -> buf1
    async_cp_b128(ldsoff + 4096, wsrc + (ks + 1) * 32);
    wait_async_le1();
    __syncthreads();
    __builtin_prefetch(arow + ks * 32 + 128, 0, 1);
    {
      v16h af = load_afrag(arow + ks * 32, c0);
#pragma unroll
      for (int c = 0; c < 4; ++c) {
        v16h bf = *(const v16h*)&Wl[0][(c * 16 + lr) * 32 + 16 * hs];
        acc[c] = wmma32(af, bf, acc[c]);
      }
    }
    __syncthreads();

    // odd step: consume buf1, prefetch ks+2 -> buf0 (except last)
    if (ks + 2 < 32) {
      async_cp_b128(ldsoff, wsrc + (ks + 2) * 32);
      wait_async_le1();
    } else {
      wait_async_le0();
    }
    __syncthreads();
    {
      v16h af = load_afrag(arow + ks * 32 + 32, c0);
#pragma unroll
      for (int c = 0; c < 4; ++c) {
        v16h bf = *(const v16h*)&Wl[1][(c * 16 + lr) * 32 + 16 * hs];
        acc[c] = wmma32(af, bf, acc[c]);
      }
    }
    __syncthreads();
  }

#pragma unroll
  for (int c = 0; c < 4; ++c) {
    int col = nt * 64 + c * 16 + lr;
    float bval = bias[col];
#pragma unroll
    for (int r = 0; r < 8; ++r) {
      int token = mt * 16 + r + 8 * hs;
      out[(size_t)token * DMODEL + col] = acc[c][r] + bval;
    }
  }
}

// ------------------------------------------------------------------ launch
extern "C" void kernel_launch(void* const* d_in, const int* in_sizes, int n_in,
                              void* d_out, int out_size, void* d_ws, size_t ws_size,
                              hipStream_t stream) {
  const float* x     = (const float*)d_in[0];   // [2,2048,1024]
  const float* inw   = (const float*)d_in[1];   // [3072,1024]
  const float* inb   = (const float*)d_in[2];   // [3072]
  const float* outw  = (const float*)d_in[3];   // [1024,1024]
  const float* outb  = (const float*)d_in[4];   // [1024]
  float* out = (float*)d_out;

  half_t* xh   = (half_t*)d_ws;                           // 4M halves
  half_t* wqkv = xh   + (size_t)NTOK * DMODEL;            // 3M
  half_t* wo   = wqkv + (size_t)3 * DMODEL * DMODEL;      // 1M
  half_t* Qb   = wo   + (size_t)DMODEL * DMODEL;          // 4M
  half_t* Kb   = Qb   + (size_t)NTOK * DMODEL;            // 4M
  half_t* Vb   = Kb   + (size_t)NTOK * DMODEL;            // 4M (transposed)
  half_t* Ah   = Vb   + (size_t)NTOK * DMODEL;            // 4M

  int n1 = NTOK * DMODEL / 4;            // x
  int n2 = 3 * DMODEL * DMODEL / 4;      // in_proj_w
  int n3 = DMODEL * DMODEL / 4;          // out_w
  cvt_f32_to_f16x4<<<(n1 + 255) / 256, 256, 0, stream>>>(x, xh, n1);
  cvt_f32_to_f16x4<<<(n2 + 255) / 256, 256, 0, stream>>>(inw, wqkv, n2);
  cvt_f32_to_f16x4<<<(n3 + 255) / 256, 256, 0, stream>>>(outw, wo, n3);

  // 32 M-slabs x 48 N-tiles = 1536 blocks of 8 waves
  qkv_gemm_kernel<<<1536, 256, 0, stream>>>(xh, wqkv, inb, Qb, Kb, Vb);

  // 2 * 16 * 128 = 4096 wave-tasks -> 512 blocks
  attn_kernel<<<512, 256, 0, stream>>>(Qb, Kb, Vb, Ah);

  // 32 M-slabs x 16 N-tiles = 512 blocks
  oproj_kernel<<<512, 256, 0, stream>>>(Ah, wo, outb, out);
}
